// SelfAttention_21045339750651
// MI455X (gfx1250) — matrix operand order repaired and checked by reading, not compiled, not run on hardware
//
#include <hip/hip_runtime.h>
#include <hip/hip_bf16.h>

// ---- problem constants (match reference) ----
#define BATCH 2
#define SEQ   2048
#define DIMSZ 1024
#define NHEAD 16
#define DHEAD 64
#define MROWS (BATCH * SEQ)          // 4096
#define SCALE_INV 0.125f             // 1/sqrt(64)

typedef _Float16 h16;
typedef __attribute__((ext_vector_type(16))) _Float16 v16h;
typedef __attribute__((ext_vector_type(8)))  _Float16 v8h;
typedef __attribute__((ext_vector_type(8)))  float    v8f;
typedef __attribute__((ext_vector_type(4)))  float    v4f;

// ---------------------------------------------------------------------------
// fp32 -> f16 conversion
// ---------------------------------------------------------------------------
__global__ void cvt_f32_to_f16(const float* __restrict__ in,
                               h16* __restrict__ out, int n) {
    int i = blockIdx.x * blockDim.x + threadIdx.x;
    if (i < n) out[i] = (h16)in[i];
}

// W (K_in, N_out) fp32 -> Wt (N_out, K_in) f16
__global__ void transpose_w_f16(const float* __restrict__ W,
                                h16* __restrict__ Wt) {
    int idx = blockIdx.x * blockDim.x + threadIdx.x;   // over DIMSZ*DIMSZ
    int k = idx & (DIMSZ - 1);
    int n = idx >> 10;
    Wt[(size_t)n * DIMSZ + k] = (h16)W[(size_t)k * DIMSZ + n];
}

// V (B,S,D) f16 -> Vt (B,H,DHEAD,S) f16
__global__ void transpose_v_heads(const h16* __restrict__ V,
                                  h16* __restrict__ Vt) {
    int idx = blockIdx.x * blockDim.x + threadIdx.x;   // over B*H*DHEAD*SEQ
    int s  = idx & (SEQ - 1);
    int d  = (idx >> 11) & (DHEAD - 1);
    int bh = idx >> 17;
    int b = bh >> 4, h = bh & 15;
    Vt[idx] = V[((size_t)(b * SEQ + s)) * DIMSZ + h * DHEAD + d];
}

// ---------------------------------------------------------------------------
// WMMA fragment helpers (wave32, 16x16x32 f16, ISA 7.12.2 layouts)
//  A (16x32): lane lo=M row; halves [0..7] -> K = hi*8..hi*8+7,
//             halves [8..15] -> K = 16+hi*8 .. 16+hi*8+7
//  B (32x16): lane lo=N col; same K order as A
// ---------------------------------------------------------------------------
__device__ __forceinline__ v16h load_a_frag_h(const h16* rowPtr, int kk, int hi) {
    const h16* p = rowPtr + kk + hi * 8;
    v8h c0 = *(const v8h*)(p);
    v8h c1 = *(const v8h*)(p + 16);
    v16h a;
#pragma unroll
    for (int i = 0; i < 8; ++i) { a[i] = c0[i]; a[i + 8] = c1[i]; }
    return a;
}

__device__ __forceinline__ v16h load_a_frag_f32(const float* rowPtr, int kk, int hi) {
    const float* p = rowPtr + kk + hi * 8;
    v4f f0 = *(const v4f*)(p);
    v4f f1 = *(const v4f*)(p + 4);
    v4f f2 = *(const v4f*)(p + 16);
    v4f f3 = *(const v4f*)(p + 20);
    v16h a;
#pragma unroll
    for (int i = 0; i < 4; ++i) {
        a[i]      = (h16)f0[i];
        a[i + 4]  = (h16)f1[i];
        a[i + 8]  = (h16)f2[i];
        a[i + 12] = (h16)f3[i];
    }
    return a;
}

__device__ __forceinline__ v16h load_b_frag_h(const h16* colPtr, int kk, int hi) {
    const h16* p = colPtr + kk + hi * 8;
    v8h c0 = *(const v8h*)(p);
    v8h c1 = *(const v8h*)(p + 16);
    v16h b;
#pragma unroll
    for (int i = 0; i < 8; ++i) { b[i] = c0[i]; b[i + 8] = c1[i]; }
    return b;
}

// ---------------------------------------------------------------------------
// GEMM: C[M,N] = A[M,K]*Bt[N,K]^T + bias ; M=4096, N=K=1024
// one wave per 16x64 tile (4 N-subtiles reuse the A fragment)
// ---------------------------------------------------------------------------
template <bool STORE_F16>
__global__ void gemm_wmma(const h16* __restrict__ A, const h16* __restrict__ Bt,
                          const float* __restrict__ bias,
                          h16* __restrict__ outh, float* __restrict__ outf) {
    const int lane = threadIdx.x;
    const int lo = lane & 15, hi = lane >> 4;
    const int m0 = blockIdx.x * 16;
    const int n0 = blockIdx.y * 64;

    const h16* arow = A + (size_t)(m0 + lo) * DIMSZ;
    const h16* bcol[4];
    v8f c[4];
#pragma unroll
    for (int t = 0; t < 4; ++t) {
        const int n = n0 + t * 16 + lo;
        bcol[t] = Bt + (size_t)n * DIMSZ;
        const float bv = bias[n];
#pragma unroll
        for (int i = 0; i < 8; ++i) c[t][i] = bv;
    }

    for (int kk = 0; kk < DIMSZ; kk += 32) {
        v16h a = load_a_frag_h(arow, kk, hi);
        __builtin_prefetch(arow + kk + 64, 0, 0);
#pragma unroll
        for (int t = 0; t < 4; ++t) {
            v16h b = load_b_frag_h(bcol[t], kk, hi);
            c[t] = __builtin_amdgcn_wmma_f32_16x16x32_f16(false, a, false, b,
                                                          (short)0, c[t], false, false);
        }
    }
#pragma unroll
    for (int t = 0; t < 4; ++t) {
#pragma unroll
        for (int r = 0; r < 8; ++r) {
            size_t o = (size_t)(m0 + hi * 8 + r) * DIMSZ + n0 + t * 16 + lo;
            if (STORE_F16) outh[o] = (h16)c[t][r];
            else           outf[o] = c[t][r];
        }
    }
}

// ---------------------------------------------------------------------------
// scores[b,h,q,k] = (Q . K) / sqrt(d); one wave per 16x64 tile, Kdim = 64
// Q fragment reused across 4 key subtiles; fully unrolled -> 8 WMMA/wave
// ---------------------------------------------------------------------------
__global__ void attn_scores(const h16* __restrict__ Q, const h16* __restrict__ Km,
                            float* __restrict__ att) {
    const int lane = threadIdx.x;
    const int lo = lane & 15, hi = lane >> 4;
    const int m0 = blockIdx.x * 16;      // query tile
    const int n0 = blockIdx.y * 64;      // key super-tile
    const int bh = blockIdx.z;
    const int b = bh >> 4, h = bh & 15;

    const h16* qrow = Q + (size_t)(b * SEQ + m0 + lo) * DIMSZ + h * DHEAD;
    const h16* kcol[4];
    v8f c[4];
#pragma unroll
    for (int t = 0; t < 4; ++t) {
        kcol[t] = Km + (size_t)(b * SEQ + n0 + t * 16 + lo) * DIMSZ + h * DHEAD;
#pragma unroll
        for (int i = 0; i < 8; ++i) c[t][i] = 0.0f;
    }

#pragma unroll
    for (int kk = 0; kk < DHEAD; kk += 32) {
        v16h a = load_a_frag_h(qrow, kk, hi);
#pragma unroll
        for (int t = 0; t < 4; ++t) {
            v16h bfr = load_b_frag_h(kcol[t], kk, hi);
            c[t] = __builtin_amdgcn_wmma_f32_16x16x32_f16(false, a, false, bfr,
                                                          (short)0, c[t], false, false);
        }
    }
#pragma unroll
    for (int t = 0; t < 4; ++t) {
#pragma unroll
        for (int r = 0; r < 8; ++r) {
            size_t o = ((size_t)bh * SEQ + (m0 + hi * 8 + r)) * SEQ + n0 + t * 16 + lo;
            att[o] = c[t][r] * SCALE_INV;
        }
    }
}

// ---------------------------------------------------------------------------
// in-place row softmax over SEQ=2048; one block per row, values kept in regs
// ---------------------------------------------------------------------------
__global__ void softmax_rows(float* __restrict__ att) {
    __shared__ float red[256];
    float* p = att + (size_t)blockIdx.x * SEQ;
    const int tid = threadIdx.x;

    float v[8];
#pragma unroll
    for (int j = 0; j < 8; ++j) v[j] = p[tid + j * 256];

    float m = v[0];
#pragma unroll
    for (int j = 1; j < 8; ++j) m = fmaxf(m, v[j]);
    red[tid] = m;
    __syncthreads();
    for (int s = 128; s > 0; s >>= 1) {
        if (tid < s) red[tid] = fmaxf(red[tid], red[tid + s]);
        __syncthreads();
    }
    const float rowmax = red[0];
    __syncthreads();

    float sum = 0.0f;
#pragma unroll
    for (int j = 0; j < 8; ++j) { v[j] = __expf(v[j] - rowmax); sum += v[j]; }
    red[tid] = sum;
    __syncthreads();
    for (int s = 128; s > 0; s >>= 1) {
        if (tid < s) red[tid] += red[tid + s];
        __syncthreads();
    }
    const float inv = 1.0f / red[0];

#pragma unroll
    for (int j = 0; j < 8; ++j) p[tid + j * 256] = v[j] * inv;
}

// ---------------------------------------------------------------------------
// context[b,q,h,:] = att[b,h,q,:] @ V[b,:,h,:]; one wave per 16x64 tile
// att rows (the 537 MB tensor) are read EXACTLY ONCE; f32->f16 in-register
// ---------------------------------------------------------------------------
__global__ void attn_context(const float* __restrict__ att,
                             const h16* __restrict__ Vt,
                             h16* __restrict__ ctx) {
    const int lane = threadIdx.x;
    const int lo = lane & 15, hi = lane >> 4;
    const int m0 = blockIdx.x * 16;      // query tile
    const int bh = blockIdx.y;
    const int b = bh >> 4, h = bh & 15;

    const float* arow = att + ((size_t)bh * SEQ + m0 + lo) * SEQ;
    const h16* vcol[4];
    v8f c[4];
#pragma unroll
    for (int t = 0; t < 4; ++t) {
        vcol[t] = Vt + ((size_t)bh * DHEAD + t * 16 + lo) * SEQ;
#pragma unroll
        for (int i = 0; i < 8; ++i) c[t][i] = 0.0f;
    }

    for (int kk = 0; kk < SEQ; kk += 32) {
        v16h a = load_a_frag_f32(arow, kk, hi);
        __builtin_prefetch(arow + kk + 64, 0, 0);
#pragma unroll
        for (int t = 0; t < 4; ++t) {
            v16h bfr = load_b_frag_h(vcol[t], kk, hi);
            c[t] = __builtin_amdgcn_wmma_f32_16x16x32_f16(false, a, false, bfr,
                                                          (short)0, c[t], false, false);
        }
    }
#pragma unroll
    for (int t = 0; t < 4; ++t) {
#pragma unroll
        for (int r = 0; r < 8; ++r) {
            size_t o = (size_t)(b * SEQ + m0 + hi * 8 + r) * DIMSZ
                     + h * DHEAD + t * 16 + lo;
            ctx[o] = (h16)c[t][r];
        }
    }
}

// ---------------------------------------------------------------------------
// out = LayerNorm(y + x) * g + b ; one block per row of DIMSZ
// ---------------------------------------------------------------------------
__global__ void residual_layernorm(const float* __restrict__ y,
                                   const float* __restrict__ x,
                                   const float* __restrict__ g,
                                   const float* __restrict__ beta,
                                   float* __restrict__ out) {
    __shared__ float rsum[256];
    __shared__ float rsq[256];
    const int tid = threadIdx.x;
    const size_t base = (size_t)blockIdx.x * DIMSZ;

    float v[4];
    float s = 0.0f, sq = 0.0f;
#pragma unroll
    for (int j = 0; j < 4; ++j) {
        int i = tid + j * 256;
        v[j] = y[base + i] + x[base + i];
        s += v[j];
        sq += v[j] * v[j];
    }
    rsum[tid] = s; rsq[tid] = sq;
    __syncthreads();
    for (int st = 128; st > 0; st >>= 1) {
        if (tid < st) { rsum[tid] += rsum[tid + st]; rsq[tid] += rsq[tid + st]; }
        __syncthreads();
    }
    const float mean = rsum[0] * (1.0f / DIMSZ);
    const float var  = rsq[0] * (1.0f / DIMSZ) - mean * mean;
    const float rstd = rsqrtf(var + 1e-5f);

#pragma unroll
    for (int j = 0; j < 4; ++j) {
        int i = tid + j * 256;
        out[base + i] = (v[j] - mean) * rstd * g[i] + beta[i];
    }
}

// ---------------------------------------------------------------------------
// host launcher
// ---------------------------------------------------------------------------
extern "C" void kernel_launch(void* const* d_in, const int* in_sizes, int n_in,
                              void* d_out, int out_size, void* d_ws, size_t ws_size,
                              hipStream_t stream) {
    (void)in_sizes; (void)n_in; (void)out_size; (void)ws_size;

    const float* x    = (const float*)d_in[0];
    const float* Wq   = (const float*)d_in[1];
    const float* bq   = (const float*)d_in[2];
    const float* Wk   = (const float*)d_in[3];
    const float* bk   = (const float*)d_in[4];
    const float* Wv   = (const float*)d_in[5];
    const float* bv   = (const float*)d_in[6];
    const float* Wo   = (const float*)d_in[7];
    const float* bo   = (const float*)d_in[8];
    const float* ln_g = (const float*)d_in[9];
    const float* ln_b = (const float*)d_in[10];

    float* out = (float*)d_out;                                  // (B,S,D)
    float* att = out + (size_t)MROWS * DIMSZ;                    // (B,H,S,S)

    // workspace carve-up (all 256B aligned by construction)
    char* ws = (char*)d_ws;
    const size_t ACT_H = (size_t)MROWS * DIMSZ * sizeof(h16);    // 8 MB
    const size_t W_H   = (size_t)DIMSZ * DIMSZ * sizeof(h16);    // 2 MB
    h16* xh   = (h16*)(ws);                 ws += ACT_H;
    h16* WtQ  = (h16*)(ws);                 ws += W_H;
    h16* WtK  = (h16*)(ws);                 ws += W_H;
    h16* WtV  = (h16*)(ws);                 ws += W_H;
    h16* WtO  = (h16*)(ws);                 ws += W_H;
    h16* Qh   = (h16*)(ws);                 ws += ACT_H;
    h16* Kh   = (h16*)(ws);                 ws += ACT_H;
    h16* Vh   = (h16*)(ws);                 ws += ACT_H;
    h16* Vt   = (h16*)(ws);                 ws += ACT_H;
    h16* ctxh = (h16*)(ws);                 ws += ACT_H;
    float* y  = (float*)(ws);               /* 16 MB */

    const int nAct = MROWS * DIMSZ;          // 4,194,304
    const int nW   = DIMSZ * DIMSZ;          // 1,048,576

    // 1) activations + weights to f16 (weights transposed to (out,in))
    cvt_f32_to_f16<<<nAct / 256, 256, 0, stream>>>(x, xh, nAct);
    transpose_w_f16<<<nW / 256, 256, 0, stream>>>(Wq, WtQ);
    transpose_w_f16<<<nW / 256, 256, 0, stream>>>(Wk, WtK);
    transpose_w_f16<<<nW / 256, 256, 0, stream>>>(Wv, WtV);
    transpose_w_f16<<<nW / 256, 256, 0, stream>>>(Wo, WtO);

    // 2) QKV projections (WMMA, 16x64 per wave)
    dim3 ggrid(MROWS / 16, DIMSZ / 64);
    gemm_wmma<true><<<ggrid, 32, 0, stream>>>(xh, WtQ, bq, Qh, nullptr);
    gemm_wmma<true><<<ggrid, 32, 0, stream>>>(xh, WtK, bk, Kh, nullptr);
    gemm_wmma<true><<<ggrid, 32, 0, stream>>>(xh, WtV, bv, Vh, nullptr);

    // 3) per-head transpose of V for the context GEMM
    transpose_v_heads<<<nAct / 256, 256, 0, stream>>>(Vh, Vt);

    // 4) raw scores straight into the att output region (WMMA, 16x64/wave)
    dim3 sgrid(SEQ / 16, SEQ / 64, BATCH * NHEAD);
    attn_scores<<<sgrid, 32, 0, stream>>>(Qh, Kh, att);

    // 5) in-place softmax (single read + single write of the 537 MB tensor)
    softmax_rows<<<BATCH * NHEAD * SEQ, 256, 0, stream>>>(att);

    // 6) context = att @ V (WMMA, att read exactly once)
    dim3 cgrid(SEQ / 16, BATCH * NHEAD);
    attn_context<<<cgrid, 32, 0, stream>>>(att, Vt, ctxh);

    // 7) output projection (WMMA, f32 store)
    gemm_wmma<false><<<ggrid, 32, 0, stream>>>(ctxh, WtO, bo, nullptr, y);

    // 8) residual + LayerNorm into the out output region
    residual_layernorm<<<MROWS, 256, 0, stream>>>(y, x, ln_g, ln_b, out);
}
